// EdgeConvResnetBlock_73521250173232
// MI455X (gfx1250) — matrix operand order repaired and checked by reading, not compiled, not run on hardware
//
#include <hip/hip_runtime.h>
#include <hip/hip_fp16.h>
#include <math.h>

typedef _Float16 h16;
typedef __attribute__((ext_vector_type(16))) _Float16 v16h;
typedef __attribute__((ext_vector_type(8)))  float    v8f;
typedef __attribute__((ext_vector_type(4)))  float    v4f;
typedef __attribute__((ext_vector_type(4)))  unsigned int v4u;

#define B_   4
#define C_   64
#define W_   64
#define H_   1024
#define SPB  (W_ * H_)          /* 65536 spatial elems per channel */
#define NT   64                 /* positions per block             */
#define NW   4                  /* waves per block                 */
#define TOTAL ((size_t)B_ * C_ * SPB)   /* 16,777,216 */

#define AZI_F 0.006135923151542565f
#define INC_F 0.008267349088394194f

// ---------------------------------------------------------------- helpers

__device__ __forceinline__ v8f wmma16(v16h a, v16h b, v8f c) {
  return __builtin_amdgcn_wmma_f32_16x16x32_f16(false, a, false, b,
                                                (short)0, c, false, false);
}

// A-operand fragment (16x32 f16) from LDS weight matrix stored [M=64][K=64].
// lane m (0..15) holds row m, halves K = kb..kb+15,
// kb = kt*32 + 16*(lane>=16).
__device__ __forceinline__ v16h ldA(const h16* w, int mt, int kt, int lane) {
  int m  = mt * 16 + (lane & 15);
  int kb = kt * 32 + ((lane >> 4) << 4);
  const h16* p = w + m * 64 + kb;
  v16h a;
  ((v4u*)&a)[0] = *(const v4u*)(p);
  ((v4u*)&a)[1] = *(const v4u*)(p + 8);
  return a;
}

// B-operand fragment (32x16 f16): lane holds 16 contiguous K halves at p.
__device__ __forceinline__ v16h ldB(const h16* p) {
  v16h b;
  ((v4u*)&b)[0] = *(const v4u*)(p);
  ((v4u*)&b)[1] = *(const v4u*)(p + 8);
  return b;
}

__device__ __forceinline__ void st8h(h16* p, v8f v) {
  union { h16 h[8]; v4u u; } t;
  #pragma unroll
  for (int i = 0; i < 8; ++i) t.h[i] = (h16)v[i];
  *(v4u*)p = t.u;
}

// ---------------------------------------------------------------- groupnorm

__global__ __launch_bounds__(256)
void gn_stats_f32(const float* __restrict__ x, float* __restrict__ stats) {
  __shared__ float ssum[256], ssq[256];
  int bg = blockIdx.x;                       // b*32 + g ; 2 channels contiguous
  const float* p = x + (size_t)bg * 131072;
  float s = 0.f, q = 0.f;
  for (int i = threadIdx.x; i < 131072; i += 256) {
    float v = p[i]; s += v; q += v * v;
  }
  ssum[threadIdx.x] = s; ssq[threadIdx.x] = q;
  __syncthreads();
  for (int o = 128; o > 0; o >>= 1) {
    if (threadIdx.x < o) {
      ssum[threadIdx.x] += ssum[threadIdx.x + o];
      ssq [threadIdx.x] += ssq [threadIdx.x + o];
    }
    __syncthreads();
  }
  if (threadIdx.x == 0) {
    float mean = ssum[0] * (1.0f / 131072.f);
    float var  = ssq[0]  * (1.0f / 131072.f) - mean * mean;
    stats[bg * 2]     = mean;
    stats[bg * 2 + 1] = rsqrtf(var + 1e-6f);
  }
}

__global__ __launch_bounds__(256)
void gn_stats_f16(const h16* __restrict__ x, float* __restrict__ stats) {
  __shared__ float ssum[256], ssq[256];
  int bg = blockIdx.x;
  const h16* p = x + (size_t)bg * 131072;
  float s = 0.f, q = 0.f;
  for (int i = threadIdx.x; i < 131072; i += 256) {
    float v = (float)p[i]; s += v; q += v * v;
  }
  ssum[threadIdx.x] = s; ssq[threadIdx.x] = q;
  __syncthreads();
  for (int o = 128; o > 0; o >>= 1) {
    if (threadIdx.x < o) {
      ssum[threadIdx.x] += ssum[threadIdx.x + o];
      ssq [threadIdx.x] += ssq [threadIdx.x + o];
    }
    __syncthreads();
  }
  if (threadIdx.x == 0) {
    float mean = ssum[0] * (1.0f / 131072.f);
    float var  = ssq[0]  * (1.0f / 131072.f) - mean * mean;
    stats[bg * 2]     = mean;
    stats[bg * 2 + 1] = rsqrtf(var + 1e-6f);
  }
}

__global__ __launch_bounds__(256)
void gn_apply_f32(const float* __restrict__ x, const float* __restrict__ stats,
                  const float* __restrict__ gw, const float* __restrict__ gb,
                  h16* __restrict__ out) {
  size_t stride = (size_t)gridDim.x * blockDim.x;
  for (size_t i = (size_t)blockIdx.x * blockDim.x + threadIdx.x; i < TOTAL; i += stride) {
    int c  = (int)((i >> 16) & 63);
    int bg = (int)(i >> 17);
    float m = stats[bg * 2], rs = stats[bg * 2 + 1];
    float v = (x[i] - m) * rs * gw[c] + gb[c];
    out[i] = (h16)fmaxf(v, 0.f);
  }
}

__global__ __launch_bounds__(256)
void gn_apply_f16(const h16* __restrict__ x, const float* __restrict__ stats,
                  const float* __restrict__ gw, const float* __restrict__ gb,
                  h16* __restrict__ out) {
  size_t stride = (size_t)gridDim.x * blockDim.x;
  for (size_t i = (size_t)blockIdx.x * blockDim.x + threadIdx.x; i < TOTAL; i += stride) {
    int c  = (int)((i >> 16) & 63);
    int bg = (int)(i >> 17);
    float m = stats[bg * 2], rs = stats[bg * 2 + 1];
    float v = ((float)x[i] - m) * rs * gw[c] + gb[c];
    out[i] = (h16)fmaxf(v, 0.f);
  }
}

// ---------------------------------------------------------------- edge conv

// LDS layout (bytes):
//  sW1a [64][64]h  @ 0       sW1b @ 8192     sW2 @ 16384
//  uT[64]f @ 24576  b1T @ 24832  b2T @ 25088  vsT[9][64]f @ 25344
//  sAct[3][66][64]h @ 27648   rT[3][66]f @ 52992   sStg[NW][16][64]h @ 53792
#define SMEM_BYTES 61984

template<int RESIDUAL>
__global__ __launch_bounds__(128)
void edgeconv_kernel(const h16*  __restrict__ act,   // [B][64][W][H] f16 (GN+relu)
                     const float* __restrict__ r,    // [B][1][W][H]
                     const float* __restrict__ w1,   // [64][131]
                     const float* __restrict__ b1,   // [64]
                     const float* __restrict__ w2,   // [64][64]
                     const float* __restrict__ b2,   // [64]
                     const float* __restrict__ xres, // residual input (f32)
                     h16*   __restrict__ outH,       // f16 out (RESIDUAL==0)
                     float* __restrict__ outF)       // f32 out (RESIDUAL==1)
{
  __shared__ __align__(16) unsigned char smem[SMEM_BYTES];
  h16*   sW1a = (h16*)  (smem + 0);
  h16*   sW1b = (h16*)  (smem + 8192);
  h16*   sW2  = (h16*)  (smem + 16384);
  float* uT   = (float*)(smem + 24576);
  float* b1T  = (float*)(smem + 24832);
  float* b2T  = (float*)(smem + 25088);
  float* vsT  = (float*)(smem + 25344);
  h16*   sAct = (h16*)  (smem + 27648);
  float* rT   = (float*)(smem + 52992);
  h16*   sStg = (h16*)  (smem + 53792);

  const int tid  = threadIdx.x;
  const int lane = tid & 31;
  const int wv   = tid >> 5;
  const int nlo  = lane & 15;
  const int hi   = lane >> 4;

  const int tilesPerRow = H_ / NT;                 // 16
  int bid  = blockIdx.x;
  int h0   = (bid % tilesPerRow) * NT;
  int wrow = (bid / tilesPerRow) % W_;
  int b    = bid / (tilesPerRow * W_);

  // ---- cooperative LDS fill ----
  for (int i = tid; i < 64 * 64; i += blockDim.x) {
    int o = i >> 6, k = i & 63;
    sW1a[o * 64 + k] = (h16)w1[o * 131 + k];
    sW1b[o * 64 + k] = (h16)w1[o * 131 + 64 + k];
    sW2 [o * 64 + k] = (h16)w2[o * 64 + k];
  }
  for (int o = tid; o < 64; o += blockDim.x) {
    float p0 = w1[o * 131 + 128];
    float p1 = w1[o * 131 + 129];
    float p2 = w1[o * 131 + 130];
    uT [o] = -p0;
    b1T[o] = b1[o];
    b2T[o] = b2[o];
    for (int s = 0; s < 9; ++s) {
      int sw = s / 3 - 1, sh = s % 3 - 1;
      float ca = __cosf((float)sw * AZI_F);
      float sa = __sinf((float)sw * AZI_F);
      float ci = __cosf((float)sh * INC_F);
      float si = __sinf((float)sh * INC_F);
      vsT[s * 64 + o] = ca * ci * p0 + ca * si * p1 + sa * p2;
    }
  }
  for (int i = tid; i < 3 * 66; i += blockDim.x) {
    int wr = i / 66, hl = i % 66;
    int wsrc = (wrow + wr - 1 + W_) & (W_ - 1);
    int hsrc = (h0 + hl - 1 + H_) & (H_ - 1);
    rT[i] = r[(size_t)b * SPB + wsrc * H_ + hsrc];
  }
  for (int i = tid; i < 3 * 66 * 64; i += blockDim.x) {
    int hl = i % 66;
    int c  = (i / 66) & 63;
    int wr = i / (66 * 64);
    int wsrc = (wrow + wr - 1 + W_) & (W_ - 1);
    int hsrc = (h0 + hl - 1 + H_) & (H_ - 1);
    sAct[(wr * 66 + hl) * 64 + c] =
        act[(((size_t)b * 64 + c) * W_ + wsrc) * H_ + hsrc];
  }
  __syncthreads();

  // ---- per-wave compute: 16 positions, all 64 out channels ----
  const int ploc = wv * 16 + nlo;            // local position = column n

  v16h aW1a[4][2], aW2[4][2];
  #pragma unroll
  for (int mt = 0; mt < 4; ++mt) {
    #pragma unroll
    for (int kt = 0; kt < 2; ++kt) {
      aW1a[mt][kt] = ldA(sW1a, mt, kt, lane);
      aW2 [mt][kt] = ldA(sW2 , mt, kt, lane);
    }
  }

  // base = W1b @ x_center + u*r_center + b1  (shift-invariant)
  float rc = rT[1 * 66 + ploc + 1];
  v8f base[4];
  {
    const h16* pc = sAct + (1 * 66 + ploc + 1) * 64;
    v16h bf0 = ldB(pc + 0 * 32 + hi * 16);
    v16h bf1 = ldB(pc + 1 * 32 + hi * 16);
    #pragma unroll
    for (int mt = 0; mt < 4; ++mt) {
      v8f c = {};
      c = wmma16(ldA(sW1b, mt, 0, lane), bf0, c);
      c = wmma16(ldA(sW1b, mt, 1, lane), bf1, c);
      int moff = mt * 16 + hi * 8;
      v4f u0 = *(const v4f*)(uT  + moff);
      v4f u1 = *(const v4f*)(uT  + moff + 4);
      v4f g0 = *(const v4f*)(b1T + moff);
      v4f g1 = *(const v4f*)(b1T + moff + 4);
      #pragma unroll
      for (int v = 0; v < 4; ++v) {
        c[v]     += u0[v] * rc + g0[v];
        c[v + 4] += u1[v] * rc + g1[v];
      }
      base[mt] = c;
    }
  }

  v8f best[4];
  #pragma unroll
  for (int mt = 0; mt < 4; ++mt) {
    #pragma unroll
    for (int v = 0; v < 8; ++v) best[mt][v] = -3.0e38f;
  }

  h16* stg = sStg + wv * 16 * 64;

  for (int s = 0; s < 9; ++s) {
    int sw = s / 3 - 1, sh = s % 3 - 1;
    int wr = 1 - sw;
    int hl = ploc + 1 - sh;                       // in [0, NT+1]
    const h16* ps = sAct + (wr * 66 + hl) * 64;
    float rs = rT[wr * 66 + hl];
    v16h bf0 = ldB(ps + 0 * 32 + hi * 16);
    v16h bf1 = ldB(ps + 1 * 32 + hi * 16);

    // GEMM1 (+base) + rank-1 pe term + ReLU -> stage as f16 [pos][ch]
    #pragma unroll
    for (int mt = 0; mt < 4; ++mt) {
      v8f c = base[mt];
      c = wmma16(aW1a[mt][0], bf0, c);
      c = wmma16(aW1a[mt][1], bf1, c);
      int moff = mt * 16 + hi * 8;
      v4f vs0 = *(const v4f*)(vsT + s * 64 + moff);
      v4f vs1 = *(const v4f*)(vsT + s * 64 + moff + 4);
      #pragma unroll
      for (int v = 0; v < 4; ++v) {
        c[v]     = fmaxf(c[v]     + vs0[v] * rs, 0.f);
        c[v + 4] = fmaxf(c[v + 4] + vs1[v] * rs, 0.f);
      }
      st8h(stg + nlo * 64 + moff, c);
    }

    // GEMM2: W2 @ h ; same-wave LDS ops are in order, no barrier needed.
    v16h hb0 = ldB(stg + nlo * 64 + 0 * 32 + hi * 16);
    v16h hb1 = ldB(stg + nlo * 64 + 1 * 32 + hi * 16);
    #pragma unroll
    for (int mt = 0; mt < 4; ++mt) {
      v8f c = {};
      c = wmma16(aW2[mt][0], hb0, c);
      c = wmma16(aW2[mt][1], hb1, c);
      #pragma unroll
      for (int v = 0; v < 8; ++v) best[mt][v] = fmaxf(best[mt][v], c[v]);
    }
  }

  // epilogue: +b2, then either store f16 intermediate or residual f32 out
  int hpos = h0 + ploc;
  #pragma unroll
  for (int mt = 0; mt < 4; ++mt) {
    int moff = mt * 16 + hi * 8;
    v4f bb0 = *(const v4f*)(b2T + moff);
    v4f bb1 = *(const v4f*)(b2T + moff + 4);
    #pragma unroll
    for (int v = 0; v < 8; ++v) {
      float val = best[mt][v] + (v < 4 ? bb0[v] : bb1[v - 4]);
      int ch = moff + v;
      size_t gi = (((size_t)b * 64 + ch) * W_ + wrow) * H_ + hpos;
      if (RESIDUAL) outF[gi] = xres[gi] + val;
      else          outH[gi] = (h16)val;
    }
  }
}

// ---------------------------------------------------------------- launch

extern "C" void kernel_launch(void* const* d_in, const int* in_sizes, int n_in,
                              void* d_out, int out_size, void* d_ws, size_t ws_size,
                              hipStream_t stream) {
  (void)in_sizes; (void)n_in; (void)out_size; (void)ws_size;
  const float* x    = (const float*)d_in[0];
  const float* r    = (const float*)d_in[1];
  const float* n1w  = (const float*)d_in[2];
  const float* n1b  = (const float*)d_in[3];
  const float* c1w1 = (const float*)d_in[4];
  const float* c1b1 = (const float*)d_in[5];
  const float* c1w2 = (const float*)d_in[6];
  const float* c1b2 = (const float*)d_in[7];
  const float* n2w  = (const float*)d_in[8];
  const float* n2b  = (const float*)d_in[9];
  const float* c2w1 = (const float*)d_in[10];
  const float* c2b1 = (const float*)d_in[11];
  const float* c2w2 = (const float*)d_in[12];
  const float* c2b2 = (const float*)d_in[13];

  h16*   hAct = (h16*)d_ws;                              // 32 MiB, reused
  h16*   e1   = (h16*)((char*)d_ws + 33554432);          // 32 MiB
  float* st   = (float*)((char*)d_ws + 67108864);        // 128 (mean,rstd) pairs

  const int ecBlocks = B_ * W_ * (H_ / NT);              // 4096

  // h1 = relu(GN1(x)) -> f16
  gn_stats_f32<<<128, 256, 0, stream>>>(x, st);
  gn_apply_f32<<<8192, 256, 0, stream>>>(x, st, n1w, n1b, hAct);
  // e1 = edge_conv(h1)
  edgeconv_kernel<0><<<ecBlocks, 128, 0, stream>>>(hAct, r, c1w1, c1b1, c1w2, c1b2,
                                                   nullptr, e1, nullptr);
  // h2 = relu(GN2(e1)) -> f16 (reuse hAct)
  gn_stats_f16<<<128, 256, 0, stream>>>(e1, st);
  gn_apply_f16<<<8192, 256, 0, stream>>>(e1, st, n2w, n2b, hAct);
  // out = x + edge_conv(h2)
  edgeconv_kernel<1><<<ecBlocks, 128, 0, stream>>>(hAct, r, c2w1, c2b1, c2w2, c2b2,
                                                   x, nullptr, (float*)d_out);
}
